// MultiHopAttention_70918499992345
// MI455X (gfx1250) — compile-verified, matrix operand order checked by reading, not run on hardware
//
#include <hip/hip_runtime.h>
#include <math.h>

#define B_   32
#define S_   2048
#define H_   1024
#define BH   (B_ * H_)
#define ROWS 32
#define PAD  8
#define HP   (H_ + PAD)

typedef __attribute__((ext_vector_type(16))) __bf16 v16bf;
typedef __attribute__((ext_vector_type(8)))  float  v8f;

union Frag { v16bf v; unsigned u[8]; };

// ---------------------------------------------------------------------------
// mean over S; u0 = mean; m0 = mean * q
// ---------------------------------------------------------------------------
__global__ __launch_bounds__(256) void mha_mean_kernel(const float* __restrict__ X,
                                                       const float* __restrict__ q,
                                                       float* __restrict__ u0,
                                                       float* __restrict__ m0)
{
    const int b = blockIdx.y;
    const int h = blockIdx.x * 256 + threadIdx.x;
    const float* Xb = X + (size_t)b * S_ * H_;
    float acc = 0.f;
    for (int s = 0; s < S_; ++s)
        acc += Xb[(size_t)s * H_ + h];
    const float mean = acc * (1.0f / (float)S_);
    u0[b * H_ + h] = mean;
    m0[b * H_ + h] = mean * q[b * H_ + h];
}

// ---------------------------------------------------------------------------
// m += u_prev * q
// ---------------------------------------------------------------------------
__global__ __launch_bounds__(256) void mha_update_m_kernel(float* __restrict__ m,
                                                           const float* __restrict__ uprev,
                                                           const float* __restrict__ q)
{
    const int i = blockIdx.x * 256 + threadIdx.x;
    m[i] += uprev[i] * q[i];
}

// ---------------------------------------------------------------------------
// W[H,H] fp32 (row = reduction h, col = output k) -> transposed bf16 Wt[col][k]
// so K-pairs for a fixed output column are contiguous dwords.
// ---------------------------------------------------------------------------
__global__ __launch_bounds__(256) void mha_convertW_kernel(const float* __restrict__ W,
                                                           __bf16* __restrict__ Wt)
{
    const int idx = blockIdx.x * 256 + threadIdx.x;
    const int r = idx >> 10;          // h (reduction)
    const int c = idx & (H_ - 1);     // k (output col)
    Wt[(size_t)c * H_ + r] = (__bf16)W[idx];
}

// ---------------------------------------------------------------------------
// t[b,k] = tanh( sum_h m[b,h] * Wm[h,k] ) * Wh[k]
// ---------------------------------------------------------------------------
__global__ __launch_bounds__(256) void mha_t_kernel(const float* __restrict__ m,
                                                    const float* __restrict__ Wm,
                                                    const float* __restrict__ Wh,
                                                    float* __restrict__ t)
{
    __shared__ float ms[H_];
    const int b = blockIdx.y;
    const int k = blockIdx.x * 256 + threadIdx.x;
    for (int i = threadIdx.x; i < H_; i += 256)
        ms[i] = m[b * H_ + i];
    __syncthreads();
    float acc = 0.f;
    for (int h = 0; h < H_; ++h)
        acc += ms[h] * Wm[h * H_ + k];
    t[b * H_ + k] = tanhf(acc) * Wh[k];
}

// ---------------------------------------------------------------------------
// Fused WMMA GEMM:  U[b,s] += sum_k tanh( (X@W)[b,s,k] ) * t[b,k]
// Block: 256 threads = 8 waves, handles 32 s-rows of one batch.
// A (16x32 bf16) staged in LDS; B from transposed bf16 W (L2-resident).
// Each wave: 8 column tiles, sharing one B fragment across two M sub-tiles.
// ---------------------------------------------------------------------------
__global__ __launch_bounds__(256) void mha_gemm_kernel(const float* __restrict__ X,
                                                       const __bf16* __restrict__ Wt,
                                                       const float* __restrict__ t,
                                                       float* __restrict__ U)
{
    __shared__ __bf16 As[ROWS * HP];

    const int tile = blockIdx.x;
    const int b    = tile / (S_ / ROWS);
    const int s0   = (tile % (S_ / ROWS)) * ROWS;
    const float* Xrow = X + ((size_t)b * S_ + s0) * H_;

    // stage 32 rows of X as bf16 into LDS (row-major, padded pitch)
    for (int i = threadIdx.x; i < ROWS * H_; i += 256) {
        const int r = i >> 10;
        const int c = i & (H_ - 1);
        As[r * HP + c] = (__bf16)Xrow[i];
    }
    __syncthreads();

    const int wave = threadIdx.x >> 5;
    const int lane = threadIdx.x & 31;
    const int half = lane >> 4;   // 0: K{0-7,16-23}, rows 0-7 of C; 1: K{8-15,24-31}, rows 8-15
    const int l15  = lane & 15;

    float racc0[8], racc1[8];
#pragma unroll
    for (int j = 0; j < 8; ++j) { racc0[j] = 0.f; racc1[j] = 0.f; }

    for (int nt = wave; nt < H_ / 16; nt += 8) {
        const int col = nt * 16 + l15;
        const __bf16* wcol = Wt + (size_t)col * H_;
        v8f acc0 = {};
        v8f acc1 = {};
        for (int kk = 0; kk < H_; kk += 32) {
            Frag af0, af1, bf;
#pragma unroll
            for (int j = 0; j < 8; ++j) {
                const int kb = ((j < 4) ? 2 * j : 16 + 2 * (j - 4)) + half * 8;
                bf.u[j]  = *(const unsigned*)(wcol + kk + kb);
                af0.u[j] = *(const unsigned*)(&As[l15 * HP + kk + kb]);
                af1.u[j] = *(const unsigned*)(&As[(16 + l15) * HP + kk + kb]);
            }
            acc0 = __builtin_amdgcn_wmma_f32_16x16x32_bf16(false, af0.v, false, bf.v,
                                                           (short)0, acc0, false, false);
            acc1 = __builtin_amdgcn_wmma_f32_16x16x32_bf16(false, af1.v, false, bf.v,
                                                           (short)0, acc1, false, false);
        }
        const float tc = t[b * H_ + col];
#pragma unroll
        for (int j = 0; j < 8; ++j) {
            racc0[j] += tanhf(acc0[j]) * tc;
            racc1[j] += tanhf(acc1[j]) * tc;
        }
    }

    // reduce over the 16 N-columns (lanes within each half-wave)
#pragma unroll
    for (int msk = 1; msk < 16; msk <<= 1) {
#pragma unroll
        for (int j = 0; j < 8; ++j) {
            racc0[j] += __shfl_xor(racc0[j], msk, 32);
            racc1[j] += __shfl_xor(racc1[j], msk, 32);
        }
    }

    if (l15 == 0) {  // lane 0 (rows 0-7) and lane 16 (rows 8-15) of each wave
        float* ub = U + (size_t)b * S_ + s0;
#pragma unroll
        for (int j = 0; j < 8; ++j) {
            atomicAdd(&ub[half * 8 + j],      racc0[j]);
            atomicAdd(&ub[16 + half * 8 + j], racc1[j]);
        }
    }
}

// ---------------------------------------------------------------------------
// softmax over S (in place), one block per batch row
// ---------------------------------------------------------------------------
__global__ __launch_bounds__(256) void mha_softmax_kernel(float* __restrict__ U)
{
    __shared__ float red[256];
    const int b = blockIdx.x;
    float* u = U + (size_t)b * S_;
    const int tid = threadIdx.x;

    float mx = -3.4e38f;
    for (int s = tid; s < S_; s += 256) mx = fmaxf(mx, u[s]);
    red[tid] = mx;
    __syncthreads();
    for (int off = 128; off > 0; off >>= 1) {
        if (tid < off) red[tid] = fmaxf(red[tid], red[tid + off]);
        __syncthreads();
    }
    const float gmax = red[0];
    __syncthreads();

    float sum = 0.f;
    for (int s = tid; s < S_; s += 256) sum += __expf(u[s] - gmax);
    red[tid] = sum;
    __syncthreads();
    for (int off = 128; off > 0; off >>= 1) {
        if (tid < off) red[tid] += red[tid + off];
        __syncthreads();
    }
    const float inv = 1.0f / red[0];

    for (int s = tid; s < S_; s += 256) u[s] = __expf(u[s] - gmax) * inv;
}

// ---------------------------------------------------------------------------
// u[b,h] = sum_s alpha[b,s] * X[b,s,h]
// ---------------------------------------------------------------------------
__global__ __launch_bounds__(256) void mha_attend_kernel(const float* __restrict__ X,
                                                         const float* __restrict__ alpha,
                                                         float* __restrict__ uout)
{
    __shared__ float al[S_];
    const int b = blockIdx.y;
    const int h = blockIdx.x * 256 + threadIdx.x;
    for (int i = threadIdx.x; i < S_; i += 256)
        al[i] = alpha[(size_t)b * S_ + i];
    __syncthreads();

    const float* Xb = X + (size_t)b * S_ * H_;
    float acc = 0.f;
    for (int s = 0; s < S_; ++s)
        acc += al[s] * Xb[(size_t)s * H_ + h];
    uout[b * H_ + h] = acc;
}

// ---------------------------------------------------------------------------
extern "C" void kernel_launch(void* const* d_in, const int* in_sizes, int n_in,
                              void* d_out, int out_size, void* d_ws, size_t ws_size,
                              hipStream_t stream)
{
    (void)in_sizes; (void)n_in; (void)out_size; (void)ws_size;

    const float* X = (const float*)d_in[0];
    const float* q = (const float*)d_in[1];
    const float* W[3]  = { (const float*)d_in[2], (const float*)d_in[5], (const float*)d_in[8]  };
    const float* Wm[3] = { (const float*)d_in[3], (const float*)d_in[6], (const float*)d_in[9]  };
    const float* Wh[3] = { (const float*)d_in[4], (const float*)d_in[7], (const float*)d_in[10] };
    float* out = (float*)d_out;   // [u0 | u1 | u2 | u3], each B*H floats

    // workspace layout
    float*  m   = (float*)d_ws;                                  // B*H
    float*  t   = m + BH;                                        // B*H
    float*  U   = t + BH;                                        // B*S (logits -> alpha)
    __bf16* Wbf = (__bf16*)((char*)d_ws + (size_t)(2 * BH + B_ * S_) * sizeof(float)); // H*H bf16

    const dim3 bh_grid(H_ / 256, B_);

    // u0 = mean(X, axis=S); m0 = u0 * q
    mha_mean_kernel<<<bh_grid, 256, 0, stream>>>(X, q, out, m);

    for (int hop = 0; hop < 3; ++hop) {
        if (hop > 0)
            mha_update_m_kernel<<<BH / 256, 256, 0, stream>>>(m, out + (size_t)(hop - 1) * BH, q);

        mha_convertW_kernel<<<(H_ * H_) / 256, 256, 0, stream>>>(W[hop], Wbf);
        mha_t_kernel<<<bh_grid, 256, 0, stream>>>(m, Wm[hop], Wh[hop], t);

        hipMemsetAsync(U, 0, (size_t)B_ * S_ * sizeof(float), stream);
        mha_gemm_kernel<<<(B_ * S_) / ROWS, 256, 0, stream>>>(X, Wbf, t, U);

        mha_softmax_kernel<<<B_, 256, 0, stream>>>(U);
        mha_attend_kernel<<<bh_grid, 256, 0, stream>>>(X, U, out + (size_t)(hop + 1) * BH);
    }
}
